// PERMAGNNTransformer_747324310167
// MI455X (gfx1250) — matrix operand order repaired and checked by reading, not compiled, not run on hardware
//
#include <hip/hip_runtime.h>
#include <hip/hip_bf16.h>
#include <math.h>

// ---------------- model constants ----------------
#define N_NODES  20000
#define E_RAW    160000
#define E_TOT    180000      // edges + self loops
#define IN_DIM   512
#define EMB      160
#define HID      240
#define GAT_H    8
#define GAT_C    30
#define TR_L     6
#define FF       1024
#define N_TOPO   4

typedef __attribute__((ext_vector_type(16))) _Float16 v16h;
typedef __attribute__((ext_vector_type(8)))  float    v8f;

// ---------------- helpers ----------------
__device__ __forceinline__ float act_apply(float x, int act) {
  switch (act) {
    case 1: return x > 0.f ? x : 0.f;                          // relu
    case 2: return tanhf(x);                                    // tanh
    case 3: return 0.5f * x * (1.f + erff(x * 0.70710678118654752f)); // exact gelu
    case 4: return 1.f / (1.f + expf(-x));                      // sigmoid
  }
  return x;
}

// order-preserving float<->uint for atomicMax on floats
__device__ __forceinline__ unsigned fenc(float f) {
  unsigned u = __float_as_uint(f);
  return (u & 0x80000000u) ? ~u : (u | 0x80000000u);
}
__device__ __forceinline__ float fdec(unsigned e) {
  return (e & 0x80000000u) ? __uint_as_float(e ^ 0x80000000u) : __uint_as_float(~e);
}
#define FENC_NEG_INF 0x007FFFFFu   // fenc(-inf)

__device__ __forceinline__ int e_src(const int* s, int e) { return (e < E_RAW) ? s[e] : (e - E_RAW); }
__device__ __forceinline__ int e_dst(const int* d, int e) { return (e < E_RAW) ? d[e] : (e - E_RAW); }

// ---------------- WMMA fragment loaders (f32 -> f16 on the fly) ----------------
// A-matrix 16x32 f16 layout: lane L: M=L&15, group g=L>>4;
// elem (v,h): K = (v>>2)*16 + g*8 + (v&3)*2 + h
__device__ __forceinline__ v16h load_a_frag(const float* A, int lda, int row, int kbase,
                                            int K, bool rowOK) {
  v16h a;
  int g = (threadIdx.x >> 4) & 1;
#pragma unroll
  for (int v = 0; v < 8; ++v) {
#pragma unroll
    for (int hh = 0; hh < 2; ++hh) {
      int kk = kbase + ((v >> 2) << 4) + (g << 3) + ((v & 3) << 1) + hh;
      float f = (rowOK && kk < K) ? A[(size_t)row * lda + kk] : 0.f;
      a[v * 2 + hh] = (_Float16)f;
    }
  }
  return a;
}
// B-matrix 32x16 f16 layout: lane L: N=L&15, group g=L>>4;
// elem (v,h): K = (v>>2)*16 + g*8 + (v&3)*2 + h
__device__ __forceinline__ v16h load_b_frag(const float* B, int ldb, int col, int kbase,
                                            int K, bool colOK) {
  v16h b;
  int g = (threadIdx.x >> 4) & 1;
#pragma unroll
  for (int v = 0; v < 8; ++v) {
#pragma unroll
    for (int hh = 0; hh < 2; ++hh) {
      int kk = kbase + ((v >> 2) << 4) + (g << 3) + ((v & 3) << 1) + hh;
      float f = (colOK && kk < K) ? B[(size_t)kk * ldb + col] : 0.f;
      b[v * 2 + hh] = (_Float16)f;
    }
  }
  return b;
}

// ---------------- generic WMMA GEMM: C = act(A@B + bias) (+= if beta) ----------------
// one wave per block; wave computes a 16(M) x 64(N) strip of C
__global__ __launch_bounds__(32) void gemm_wmma(const float* __restrict__ A, int lda,
                                                const float* __restrict__ B, int ldb,
                                                const float* __restrict__ bias,
                                                float* __restrict__ C, int ldc,
                                                int M, int K, int N, int act, int beta) {
  int lane = threadIdx.x;
  int tm  = blockIdx.y * 16;
  int tn0 = blockIdx.x * 64;
  int arow = tm + (lane & 15);
  bool rowOK = arow < M;

  v8f acc[4] = {};
  for (int k0 = 0; k0 < K; k0 += 32) {
    if (rowOK && (k0 + 32) < K)
      __builtin_prefetch(&A[(size_t)arow * lda + k0 + 32], 0, 1);
    v16h a = load_a_frag(A, lda, arow, k0, K, rowOK);
#pragma unroll
    for (int t = 0; t < 4; ++t) {
      int colb = tn0 + t * 16;
      if (colb >= N) continue;                 // uniform across the wave
      int col = colb + (lane & 15);
      v16h b = load_b_frag(B, ldb, col, k0, K, col < N);
      acc[t] = __builtin_amdgcn_wmma_f32_16x16x32_f16(
          false, a, false, b, (short)0, acc[t], false, false);
    }
  }

  int cg = lane >> 4;
#pragma unroll
  for (int t = 0; t < 4; ++t) {
    int colb = tn0 + t * 16;
    if (colb >= N) break;
    int col = colb + (lane & 15);
#pragma unroll
    for (int r = 0; r < 8; ++r) {
      int row = tm + r + cg * 8;
      if (row < M && col < N) {
        float v = acc[t][r] + (bias ? bias[col] : 0.f);
        v = act_apply(v, act);
        size_t idx = (size_t)row * ldc + col;
        if (beta) C[idx] += v; else C[idx] = v;
      }
    }
  }
}

// ---------------- small utility kernels ----------------
__global__ void fill_f32(float* p, float v, long n) {
  long i = (long)blockIdx.x * blockDim.x + threadIdx.x;
  if (i < n) p[i] = v;
}
__global__ void fill_u32(unsigned* p, unsigned v, long n) {
  long i = (long)blockIdx.x * blockDim.x + threadIdx.x;
  if (i < n) p[i] = v;
}

// degrees (in-degree over dst incl. self loops)
__global__ void deg_kernel(const int* esrc, const int* edst, float* deg) {
  int e = blockIdx.x * blockDim.x + threadIdx.x;
  if (e >= E_TOT) return;
  atomicAdd(&deg[e_dst(edst, e)], 1.f);
}
__global__ void dinv_kernel(const float* deg, float* dinv) {
  int n = blockIdx.x * blockDim.x + threadIdx.x;
  if (n >= N_NODES) return;
  float d = deg[n];
  dinv[n] = d > 0.f ? rsqrtf(fmaxf(d, 1e-12f)) : 0.f;
}

// GCN scatter: agg[dst,d] += h[src,d] * dinv[src]*dinv[dst]
__global__ void gcn_agg_kernel(const int* esrc, const int* edst,
                               const float* __restrict__ h, const float* __restrict__ dinv,
                               float* __restrict__ agg) {
  long i = (long)blockIdx.x * blockDim.x + threadIdx.x;
  if (i >= (long)E_TOT * HID) return;
  int e = (int)(i / HID), d = (int)(i % HID);
  int s = e_src(esrc, e), t = e_dst(edst, e);
  atomicAdd(&agg[(size_t)t * HID + d], h[(size_t)s * HID + d] * dinv[s] * dinv[t]);
}

__global__ void bias_act_kernel(const float* in, const float* bias, float* out,
                                long n, int D, int act) {
  long i = (long)blockIdx.x * blockDim.x + threadIdx.x;
  if (i >= n) return;
  out[i] = act_apply(in[i] + bias[i % D], act);
}

// GAT attention scores per node/head
__global__ void gat_score_kernel(const float* __restrict__ hg, const float* __restrict__ a_s,
                                 const float* __restrict__ a_d, float* es, float* ed) {
  int i = blockIdx.x * blockDim.x + threadIdx.x;
  if (i >= N_NODES * GAT_H) return;
  int n = i / GAT_H, hd = i % GAT_H;
  float s1 = 0.f, s2 = 0.f;
  const float* hp = hg + (size_t)n * HID + hd * GAT_C;
#pragma unroll
  for (int c = 0; c < GAT_C; ++c) {
    float v = hp[c];
    s1 += v * a_s[hd * GAT_C + c];
    s2 += v * a_d[hd * GAT_C + c];
  }
  es[i] = s1; ed[i] = s2;
}

// per-edge leaky-relu score + segment max (encoded atomicMax)
__global__ void edge_max_kernel(const int* esrc, const int* edst,
                                const float* es, const float* ed,
                                float* eact, unsigned* mx) {
  long i = (long)blockIdx.x * blockDim.x + threadIdx.x;
  if (i >= (long)E_TOT * GAT_H) return;
  int e = (int)(i / GAT_H), hd = (int)(i % GAT_H);
  int s = e_src(esrc, e), t = e_dst(edst, e);
  float v = es[s * GAT_H + hd] + ed[t * GAT_H + hd];
  v = v > 0.f ? v : 0.2f * v;                    // leaky_relu(0.2)
  eact[i] = v;
  atomicMax(&mx[t * GAT_H + hd], fenc(v));
}

// exp(e - max) + segment sum
__global__ void edge_exp_kernel(const int* edst, float* eact, const unsigned* mx,
                                float* ssum) {
  long i = (long)blockIdx.x * blockDim.x + threadIdx.x;
  if (i >= (long)E_TOT * GAT_H) return;
  int e = (int)(i / GAT_H), hd = (int)(i % GAT_H);
  int t = e_dst(edst, e);
  float m = fdec(mx[t * GAT_H + hd]);
  float ex = expf(eact[i] - m);
  eact[i] = ex;
  atomicAdd(&ssum[t * GAT_H + hd], ex);
}

// weighted aggregate: agg[dst,d] += hg[src,d] * alpha(e, head(d))
__global__ void edge_gat_kernel(const int* esrc, const int* edst,
                                const float* __restrict__ hg, const float* __restrict__ eact,
                                const float* __restrict__ ssum, float* __restrict__ agg) {
  long i = (long)blockIdx.x * blockDim.x + threadIdx.x;
  if (i >= (long)E_TOT * HID) return;
  int e = (int)(i / HID), d = (int)(i % HID);
  int hd = d / GAT_C;
  int s = e_src(esrc, e), t = e_dst(edst, e);
  float alpha = eact[(size_t)e * GAT_H + hd] / ssum[t * GAT_H + hd];
  atomicAdd(&agg[(size_t)t * HID + d], hg[(size_t)s * HID + d] * alpha);
}

// LayerNorm over D=240; one wave32 per row. out = LN(x (+res) (+preb)) * g + b
__global__ __launch_bounds__(256) void ln_kernel(const float* __restrict__ x,
                                                 const float* __restrict__ res,
                                                 const float* __restrict__ preb,
                                                 const float* __restrict__ g,
                                                 const float* __restrict__ b,
                                                 float* __restrict__ out, int ldo, int rows) {
  int wave = (blockIdx.x * blockDim.x + threadIdx.x) >> 5;
  int lane = threadIdx.x & 31;
  if (wave >= rows) return;
  float vals[8];
  float s = 0.f;
#pragma unroll
  for (int i = 0; i < 8; ++i) {
    int d = lane + i * 32;
    float v = 0.f;
    if (d < HID) {
      v = x[(size_t)wave * HID + d];
      if (res)  v += res[(size_t)wave * HID + d];
      if (preb) v += preb[d];
    }
    vals[i] = v; s += v;
  }
#pragma unroll
  for (int o = 16; o > 0; o >>= 1) s += __shfl_xor(s, o, 32);
  float mean = s / (float)HID;
  float vs = 0.f;
#pragma unroll
  for (int i = 0; i < 8; ++i) {
    int d = lane + i * 32;
    if (d < HID) { float dv = vals[i] - mean; vs += dv * dv; }
  }
#pragma unroll
  for (int o = 16; o > 0; o >>= 1) vs += __shfl_xor(vs, o, 32);
  float inv = rsqrtf(vs / (float)HID + 1e-5f);
#pragma unroll
  for (int i = 0; i < 8; ++i) {
    int d = lane + i * 32;
    if (d < HID)
      out[(size_t)wave * ldo + d] = (vals[i] - mean) * inv * g[d] + b[d];
  }
}

// 5-token / 5-head cross attention (hd=32); thread per (node, head, qtok)
__global__ void attn5_kernel(const float* __restrict__ qkv, float* __restrict__ attno) {
  int i = blockIdx.x * blockDim.x + threadIdx.x;
  if (i >= N_NODES * 25) return;
  int n = i / 25, r = i % 25, hd = r / 5, q = r % 5;
  const float* qp = qkv + ((size_t)(n * 5 + q)) * 480 + hd * 32;
  float sc[5], mx = -1e30f;
#pragma unroll
  for (int kt = 0; kt < 5; ++kt) {
    const float* kp = qkv + ((size_t)(n * 5 + kt)) * 480 + 160 + hd * 32;
    float s = 0.f;
#pragma unroll
    for (int d = 0; d < 32; ++d) s += qp[d] * kp[d];
    s *= 0.17677669529663687f;                  // 1/sqrt(32)
    sc[kt] = s; mx = fmaxf(mx, s);
  }
  float se = 0.f;
#pragma unroll
  for (int kt = 0; kt < 5; ++kt) { sc[kt] = expf(sc[kt] - mx); se += sc[kt]; }
  float inv = 1.f / se;
  float* op = attno + ((size_t)(n * 5 + q)) * EMB + hd * 32;
  for (int d = 0; d < 32; ++d) {
    float o = 0.f;
#pragma unroll
    for (int kt = 0; kt < 5; ++kt)
      o += sc[kt] * qkv[((size_t)(n * 5 + kt)) * 480 + 320 + hd * 32 + d];
    op[d] = o * inv;
  }
}

// fuse: logits=t@w2+b2 -> softmax(4) -> h = sum_k w_k * stacked[:,k,:]
__global__ void fuse_combine_kernel(const float* __restrict__ t, const float* __restrict__ w2,
                                    const float* __restrict__ b2,
                                    const float* __restrict__ stacked, float* __restrict__ h) {
  int n = blockIdx.x * blockDim.x + threadIdx.x;
  if (n >= N_NODES) return;
  float lg[4] = {b2[0], b2[1], b2[2], b2[3]};
  for (int d = 0; d < HID; ++d) {
    float v = t[(size_t)n * HID + d];
#pragma unroll
    for (int j = 0; j < 4; ++j) lg[j] += v * w2[d * 4 + j];
  }
  float m = fmaxf(fmaxf(lg[0], lg[1]), fmaxf(lg[2], lg[3]));
  float s = 0.f;
#pragma unroll
  for (int j = 0; j < 4; ++j) { lg[j] = expf(lg[j] - m); s += lg[j]; }
  float inv = 1.f / s;
  for (int d = 0; d < HID; ++d) {
    float o = 0.f;
#pragma unroll
    for (int j = 0; j < 4; ++j) o += lg[j] * stacked[(size_t)n * 960 + j * HID + d];
    h[(size_t)n * HID + d] = o * inv;
  }
}

// heads
__global__ void wb_out_kernel(const float* __restrict__ u, const float* __restrict__ w2,
                              const float* __restrict__ b2, float* out) {
  int n = blockIdx.x * blockDim.x + threadIdx.x;
  if (n >= N_NODES) return;
  float s = b2[0];
  for (int d = 0; d < 120; ++d) s += u[(size_t)n * 120 + d] * w2[d];
  out[n] = 1.f / (1.f + expf(-s));
}
__global__ void pm_out_kernel(const float* __restrict__ v, const float* __restrict__ w2,
                              const float* __restrict__ b2, float* out) {
  int i = blockIdx.x * blockDim.x + threadIdx.x;
  if (i >= N_NODES * 5) return;
  int n = i / 5, j = i % 5;
  float s = b2[j];
  for (int d = 0; d < 120; ++d) s += v[(size_t)n * 120 + d] * w2[d * 5 + j];
  out[i] = 1.f / (1.f + expf(-s));
}

// ---------------- launch ----------------
static inline int cdiv(long a, long b) { return (int)((a + b - 1) / b); }

extern "C" void kernel_launch(void* const* d_in, const int* in_sizes, int n_in,
                              void* d_out, int out_size, void* d_ws, size_t ws_size,
                              hipStream_t stream) {
  (void)in_sizes; (void)n_in; (void)out_size; (void)ws_size;
  const float* x        = (const float*)d_in[0];
  const int*   ei       = (const int*)d_in[1];
  const float* perma_W  = (const float*)d_in[2];
  const float* perma_b  = (const float*)d_in[3];
  const float* ca_in_w  = (const float*)d_in[4];
  const float* ca_in_b  = (const float*)d_in[5];
  const float* ca_out_w = (const float*)d_in[6];
  const float* ca_out_b = (const float*)d_in[7];
  const float* style_w  = (const float*)d_in[8];
  const float* style_b  = (const float*)d_in[9];
  const float* stress_w = (const float*)d_in[10];
  const float* stress_b = (const float*)d_in[11];
  const float* gcn_w0   = (const float*)d_in[12];
  const float* gcn_b0   = (const float*)d_in[13];
  const float* gcn_w1   = (const float*)d_in[14];
  const float* gcn_b1   = (const float*)d_in[15];
  const float* gcn_w2   = (const float*)d_in[16];
  const float* gcn_b2   = (const float*)d_in[17];
  const float* gat_w    = (const float*)d_in[18];
  const float* gat_as   = (const float*)d_in[19];
  const float* gat_ad   = (const float*)d_in[20];
  const float* gat_b    = (const float*)d_in[21];
  const float* ln_g     = (const float*)d_in[22];
  const float* ln_b     = (const float*)d_in[23];
  const float* fuse_w1  = (const float*)d_in[24];
  const float* fuse_b1  = (const float*)d_in[25];
  const float* fuse_w2  = (const float*)d_in[26];
  const float* fuse_b2  = (const float*)d_in[27];
  const float* tr_in_w  = (const float*)d_in[28];
  const float* tr_in_b  = (const float*)d_in[29];
  const float* tr_out_w = (const float*)d_in[30];
  const float* tr_out_b = (const float*)d_in[31];
  const float* tr_l1_w  = (const float*)d_in[32];
  const float* tr_l1_b  = (const float*)d_in[33];
  const float* tr_l2_w  = (const float*)d_in[34];
  const float* tr_l2_b  = (const float*)d_in[35];
  const float* tr_ln1_g = (const float*)d_in[36];
  const float* tr_ln1_b = (const float*)d_in[37];
  const float* tr_ln2_g = (const float*)d_in[38];
  const float* tr_ln2_b = (const float*)d_in[39];
  const float* wb_w1    = (const float*)d_in[40];
  const float* wb_b1    = (const float*)d_in[41];
  const float* wb_w2    = (const float*)d_in[42];
  const float* wb_b2    = (const float*)d_in[43];
  const float* pm_w1    = (const float*)d_in[44];
  const float* pm_b1    = (const float*)d_in[45];
  const float* pm_w2    = (const float*)d_in[46];
  const float* pm_b2    = (const float*)d_in[47];
  float* fout = (float*)d_out;

  // ---- workspace layout (floats) ----
  float* W = (float*)d_ws;
  size_t off = 0;
  float* flat    = W + off; off += (size_t)N_NODES * 800;   // perma tokens == flat [N,800] / [5N,160]
  float* ctx     = W + off; off += (size_t)N_NODES * 480;   // [style | stress]
  float* stacked = W + off; off += (size_t)N_NODES * 960;   // [N,4,240]
  float* hbuf    = W + off; off += (size_t)N_NODES * HID;   // fused h
  float* S       = W + off;                                  // stage scratch

  auto gemm = [&](const float* A, int lda, const float* B, int ldb, const float* bias,
                  float* C, int ldc, int M, int K, int N, int act, int beta) {
    dim3 g(cdiv(N, 64), cdiv(M, 16));
    gemm_wmma<<<g, dim3(32), 0, stream>>>(A, lda, B, ldb, bias, C, ldc, M, K, N, act, beta);
  };
  const int TB = 256;

  // ---- 1. PERMA embedding: 5 projections ----
  for (int k = 0; k < 5; ++k)
    gemm(x, IN_DIM, perma_W + (size_t)k * IN_DIM * EMB, EMB, perma_b + k * EMB,
         flat + k * EMB, 800, N_NODES, IN_DIM, EMB, 0, 0);

  // ---- 2. residual cross-attention over 5 tokens ----
  float* qkv   = S;                                  // [5N,480]
  float* attno = S + (size_t)N_NODES * 5 * 480;      // [5N,160]
  gemm(flat, EMB, ca_in_w, 480, ca_in_b, qkv, 480, N_NODES * 5, EMB, 480, 0, 0);
  attn5_kernel<<<cdiv((long)N_NODES * 25, TB), TB, 0, stream>>>(qkv, attno);
  gemm(attno, EMB, ca_out_w, EMB, ca_out_b, flat, EMB, N_NODES * 5, EMB, EMB, 0, 1); // += residual

  // ---- 3. style / stress context ----
  gemm(x, IN_DIM, style_w,  HID, style_b,  ctx,       480, N_NODES, IN_DIM, HID, 1, 0);
  gemm(x, IN_DIM, stress_w, HID, stress_b, ctx + HID, 480, N_NODES, IN_DIM, HID, 1, 0);

  // ---- 4. per-topology GCN×3 + GAT + LN ----
  float*    t0   = S;
  float*    agg  = S + (size_t)N_NODES * HID;
  float*    hcur = S + (size_t)N_NODES * HID * 2;
  float*    hg   = S + (size_t)N_NODES * HID * 3;
  float*    deg  = S + (size_t)N_NODES * HID * 4;
  float*    dinv = deg + N_NODES;
  float*    es   = dinv + N_NODES;
  float*    ed   = es + (size_t)N_NODES * GAT_H;
  unsigned* mx   = (unsigned*)(ed + (size_t)N_NODES * GAT_H);
  float*    ssum = (float*)mx + (size_t)N_NODES * GAT_H;
  float*    eact = ssum + (size_t)N_NODES * GAT_H;          // [E_TOT,8]

  for (int k = 0; k < N_TOPO; ++k) {
    const int* esrc = ei + (size_t)k * 2 * E_RAW;
    const int* edst = esrc + E_RAW;

    fill_f32<<<cdiv(N_NODES, TB), TB, 0, stream>>>(deg, 0.f, N_NODES);
    deg_kernel<<<cdiv(E_TOT, TB), TB, 0, stream>>>(esrc, edst, deg);
    dinv_kernel<<<cdiv(N_NODES, TB), TB, 0, stream>>>(deg, dinv);

    // GCN layer 0 (flat 800 -> 240)
    gemm(flat, 800, gcn_w0 + (size_t)k * 800 * HID, HID, nullptr, t0, HID,
         N_NODES, 800, HID, 0, 0);
    fill_f32<<<cdiv((long)N_NODES * HID, TB), TB, 0, stream>>>(agg, 0.f, (long)N_NODES * HID);
    gcn_agg_kernel<<<cdiv((long)E_TOT * HID, TB), TB, 0, stream>>>(esrc, edst, t0, dinv, agg);
    bias_act_kernel<<<cdiv((long)N_NODES * HID, TB), TB, 0, stream>>>(
        agg, gcn_b0 + k * HID, hcur, (long)N_NODES * HID, HID, 1);

    // GCN layers 1 & 2 (240 -> 240)
    const float* w12[2] = {gcn_w1 + (size_t)k * HID * HID, gcn_w2 + (size_t)k * HID * HID};
    const float* b12[2] = {gcn_b1 + k * HID, gcn_b2 + k * HID};
    for (int l = 0; l < 2; ++l) {
      gemm(hcur, HID, w12[l], HID, nullptr, t0, HID, N_NODES, HID, HID, 0, 0);
      fill_f32<<<cdiv((long)N_NODES * HID, TB), TB, 0, stream>>>(agg, 0.f, (long)N_NODES * HID);
      gcn_agg_kernel<<<cdiv((long)E_TOT * HID, TB), TB, 0, stream>>>(esrc, edst, t0, dinv, agg);
      bias_act_kernel<<<cdiv((long)N_NODES * HID, TB), TB, 0, stream>>>(
          agg, b12[l], hcur, (long)N_NODES * HID, HID, 1);
    }

    // GAT
    gemm(hcur, HID, gat_w + (size_t)k * HID * HID, HID, nullptr, hg, HID,
         N_NODES, HID, HID, 0, 0);
    gat_score_kernel<<<cdiv((long)N_NODES * GAT_H, TB), TB, 0, stream>>>(
        hg, gat_as + k * HID, gat_ad + k * HID, es, ed);
    fill_u32<<<cdiv((long)N_NODES * GAT_H, TB), TB, 0, stream>>>(
        mx, FENC_NEG_INF, (long)N_NODES * GAT_H);
    edge_max_kernel<<<cdiv((long)E_TOT * GAT_H, TB), TB, 0, stream>>>(
        esrc, edst, es, ed, eact, mx);
    fill_f32<<<cdiv((long)N_NODES * GAT_H, TB), TB, 0, stream>>>(
        ssum, 0.f, (long)N_NODES * GAT_H);
    edge_exp_kernel<<<cdiv((long)E_TOT * GAT_H, TB), TB, 0, stream>>>(edst, eact, mx, ssum);
    fill_f32<<<cdiv((long)N_NODES * HID, TB), TB, 0, stream>>>(agg, 0.f, (long)N_NODES * HID);
    edge_gat_kernel<<<cdiv((long)E_TOT * HID, TB), TB, 0, stream>>>(
        esrc, edst, hg, eact, ssum, agg);

    // + gat_b, LayerNorm, write into stacked[:,k,:]
    ln_kernel<<<cdiv((long)N_NODES * 32, TB), TB, 0, stream>>>(
        agg, nullptr, gat_b + k * HID, ln_g + k * HID, ln_b + k * HID,
        stacked + k * HID, 960, N_NODES);
  }

  // ---- 5. fuse ----
  float* tfuse = S;
  gemm(ctx, 480, fuse_w1, HID, fuse_b1, tfuse, HID, N_NODES, 480, HID, 2, 0); // tanh
  fuse_combine_kernel<<<cdiv(N_NODES, TB), TB, 0, stream>>>(tfuse, fuse_w2, fuse_b2,
                                                            stacked, hbuf);

  // ---- 6. transformer (seq_len=1 => attn == out_proj(v_proj(h))) ----
  float* tbuf = S;
  float* abuf = S + (size_t)N_NODES * HID;
  float* f1   = S + (size_t)N_NODES * HID * 2;              // [N,1024]
  for (int l = 0; l < TR_L; ++l) {
    gemm(hbuf, HID, tr_in_w + (size_t)l * HID * 720 + 480, 720,
         tr_in_b + l * 720 + 480, tbuf, HID, N_NODES, HID, HID, 0, 0);
    gemm(tbuf, HID, tr_out_w + (size_t)l * HID * HID, HID,
         tr_out_b + l * HID, abuf, HID, N_NODES, HID, HID, 0, 0);
    ln_kernel<<<cdiv((long)N_NODES * 32, TB), TB, 0, stream>>>(
        hbuf, abuf, nullptr, tr_ln1_g + l * HID, tr_ln1_b + l * HID, hbuf, HID, N_NODES);
    gemm(hbuf, HID, tr_l1_w + (size_t)l * HID * FF, FF,
         tr_l1_b + l * FF, f1, FF, N_NODES, HID, FF, 3, 0);          // exact gelu
    gemm(f1, FF, tr_l2_w + (size_t)l * FF * HID, HID,
         tr_l2_b + l * HID, abuf, HID, N_NODES, FF, HID, 0, 0);
    ln_kernel<<<cdiv((long)N_NODES * 32, TB), TB, 0, stream>>>(
        hbuf, abuf, nullptr, tr_ln2_g + l * HID, tr_ln2_b + l * HID, hbuf, HID, N_NODES);
  }

  // ---- 7. heads ----
  float* ubuf = S;
  float* vbuf = S + (size_t)N_NODES * 120;
  gemm(hbuf, HID, wb_w1, 120, wb_b1, ubuf, 120, N_NODES, HID, 120, 1, 0);
  wb_out_kernel<<<cdiv(N_NODES, TB), TB, 0, stream>>>(ubuf, wb_w2, wb_b2, fout);
  gemm(hbuf, HID, pm_w1, 120, pm_b1, vbuf, 120, N_NODES, HID, 120, 1, 0);
  pm_out_kernel<<<cdiv((long)N_NODES * 5, TB), TB, 0, stream>>>(vbuf, pm_w2, pm_b2,
                                                                fout + N_NODES);
}
